// SpatialTemporalTransformerLayer_30700426232555
// MI455X (gfx1250) — matrix operand order repaired and checked by reading, not compiled, not run on hardware
//
#include <hip/hip_runtime.h>
#include <hip/hip_bf16.h>

#define KN   4096
#define SEQ  32
#define DM   128
#define DFF  512
#define NE   65536
#define NET  (NE + KN)          // edges + self loops = 69632

typedef __attribute__((ext_vector_type(16))) _Float16 v16h;
typedef __attribute__((ext_vector_type(8)))  _Float16 v8h;
typedef __attribute__((ext_vector_type(4)))  _Float16 v4h;
typedef __attribute__((ext_vector_type(8)))  float    v8f;
typedef __attribute__((ext_vector_type(4)))  unsigned v4u;
typedef __attribute__((ext_vector_type(8)))  int      v8i;
typedef __attribute__((ext_vector_type(4)))  int      v4i;

#define WMMA_F16(a, b, c) \
  __builtin_amdgcn_wmma_f32_16x16x32_f16(false, (a), false, (b), (short)0, (c), false, false)

// ---- Tensor Data Mover: DMA a 2D tile (row_elems x nrows, contiguous rows) into LDS ----
// data_size_code: 0=1B,1=2B,2=4B.  Issue from ONE wave only (uniform branch); EXEC ignored.
__device__ __forceinline__ void tdm_load_tile(unsigned lds_addr, const void* gptr,
                                              unsigned data_size_code, unsigned row_elems,
                                              unsigned nrows, unsigned nrows_total) {
  unsigned long long ga = (unsigned long long)(size_t)gptr;
  v4u g0;
  g0[0] = 1u;                                                   // count=1, user descriptor
  g0[1] = lds_addr;                                             // LDS byte address
  g0[2] = (unsigned)(ga & 0xffffffffu);                         // global_addr[31:0]
  g0[3] = (unsigned)((ga >> 32) & 0x01ffffffu) | (2u << 30);    // global_addr[56:32] | type=2
  v8i g1;
  g1[0] = (int)(data_size_code << 16);                          // wg_mask=0 | data_size
  g1[1] = (int)((row_elems & 0xffffu) << 16);                   // ab_addr=0 | tensor_dim0[15:0]
  g1[2] = (int)((row_elems >> 16) | ((nrows_total & 0xffffu) << 16)); // t_dim0 hi | t_dim1 lo
  g1[3] = (int)((nrows_total >> 16) | ((row_elems & 0xffffu) << 16)); // t_dim1 hi | tile_dim0
  g1[4] = (int)nrows;                                           // tile_dim1 | tile_dim2=0
  g1[5] = (int)row_elems;                                       // tensor_dim0_stride[31:0]
  g1[6] = 0;                                                    // stride0 hi | stride1 lo
  g1[7] = 0;
  v4i z4 = {};
  v8i z8 = {};
  __builtin_amdgcn_tensor_load_to_lds(g0, g1, z4, z4, z8, 0);
}

// ---- WMMA fragment loaders (contiguous k = 16*(lane/16) + s in both A and B, wave32) ----

// A: 16x32 f16 tile from row-major matrix (lda elements), rows m0..m0+15, cols k0..k0+31
__device__ __forceinline__ v16h ldA(const _Float16* A, int lda, int m0, int k0, int lane) {
  int row = m0 + (lane & 15);
  int kh  = (lane >> 4) << 4;                 // 0 or 16
  const _Float16* p = A + row * lda + k0;
  v16h a = {};
#pragma unroll
  for (int v = 0; v < 8; ++v) {
    int kk = kh + 2 * v;
    a[2 * v]     = p[kk];
    a[2 * v + 1] = p[kk + 1];
  }
  return a;
}

// B: 32x16 where B[k][n] = W[n][k], W row-major [N][K] (PyTorch h @ W.T convention).
__device__ __forceinline__ v16h ldB_nk(const _Float16* W, int ldw, int n0, int k0, int lane) {
  const _Float16* p = W + (size_t)(n0 + (lane & 15)) * ldw + k0 + ((lane >> 4) << 4);
  v16h b = {};
#pragma unroll
  for (int i = 0; i < 16; ++i) b[i] = p[i];
  return b;
}

// B: 32x16 where B[k][n] = V[k][n], V row-major [K][N] (used for probs @ V).
__device__ __forceinline__ v16h ldB_kn(const _Float16* V, int ldv, int k0, int n0, int lane) {
  int n  = n0 + (lane & 15);
  int kb = k0 + ((lane >> 4) << 4);
  v16h b = {};
#pragma unroll
  for (int v = 0; v < 8; ++v) {
    b[2 * v]     = V[(size_t)(kb + 2 * v)     * ldv + n];
    b[2 * v + 1] = V[(size_t)(kb + 2 * v + 1) * ldv + n];
  }
  return b;
}

// ---- float <-> ordered-uint for atomic max on floats ----
__device__ __forceinline__ unsigned fenc(float f) {
  unsigned u = __float_as_uint(f);
  return (u & 0x80000000u) ? ~u : (u | 0x80000000u);
}
__device__ __forceinline__ float fdec(unsigned u) {
  return __uint_as_float((u & 0x80000000u) ? (u ^ 0x80000000u) : ~u);
}

// ---- LayerNorm over last dim (128), 8 threads per row, output f16 ----
__device__ __forceinline__ void ln_rows(const float (*xb)[DM], _Float16 (*hb)[DM],
                                        const float* g, const float* bb, int t) {
  int row = t >> 3, sub = t & 7;
  float s = 0.f, s2 = 0.f;
#pragma unroll
  for (int i = 0; i < 16; ++i) {
    float v = xb[row][sub * 16 + i];
    s += v; s2 += v * v;
  }
  for (int m = 4; m; m >>= 1) { s += __shfl_xor(s, m, 8); s2 += __shfl_xor(s2, m, 8); }
  float mean = s * (1.f / DM);
  float inv  = rsqrtf(s2 * (1.f / DM) - mean * mean + 1e-5f);
#pragma unroll
  for (int i = 0; i < 16; ++i) {
    int c = sub * 16 + i;
    hb[row][c] = (_Float16)((xb[row][c] - mean) * inv * g[c] + bb[c]);
  }
}

// ============ Kernel 0: convert weights to f16, packed [Wqkv|Wout|Wff1|Wff2|Wgat] ============
__global__ void k_prep(const float* __restrict__ qkvw, const float* __restrict__ outw,
                       const float* __restrict__ ff1w, const float* __restrict__ ff2w,
                       const float* __restrict__ gatw, _Float16* __restrict__ w16) {
  int i = blockIdx.x * 256 + threadIdx.x;
  const int N0 = 49152, N1 = N0 + 16384, N2 = N1 + 65536, N3 = N2 + 65536, N4 = N3 + 16384;
  if      (i < N0) w16[i] = (_Float16)qkvw[i];
  else if (i < N1) w16[i] = (_Float16)outw[i - N0];
  else if (i < N2) w16[i] = (_Float16)ff1w[i - N1];
  else if (i < N3) w16[i] = (_Float16)ff2w[i - N2];
  else if (i < N4) w16[i] = (_Float16)gatw[i - N3];
}

// ============ Kernel 1: init output (= gat_b broadcast) and softmax accumulators ============
__global__ void k_init(float* __restrict__ out, const float* __restrict__ gat_b,
                       unsigned* __restrict__ mkey, float* __restrict__ denom) {
  int i = blockIdx.x * 256 + threadIdx.x;
  if (i < KN * SEQ * DM) out[i] = gat_b[i & (DM - 1)];
  if (i < KN * SEQ) { mkey[i] = 0u; denom[i] = 0.f; }
}

// ============ Kernel 2: fused temporal transformer layer, one node per block ============
__launch_bounds__(256)
__global__ void k_temporal(const float* __restrict__ x,
                           const float* __restrict__ ln1_g, const float* __restrict__ ln1_b,
                           const float* __restrict__ qkv_b, const float* __restrict__ out_b,
                           const float* __restrict__ ln2_g, const float* __restrict__ ln2_b,
                           const float* __restrict__ ff1_b, const float* __restrict__ ff2_b,
                           const _Float16* __restrict__ Wqkv, const _Float16* __restrict__ Wout,
                           const _Float16* __restrict__ Wff1, const _Float16* __restrict__ Wff2,
                           _Float16* __restrict__ xt) {
  __shared__ float    xbuf[SEQ][DM];                 // 16 KB, residual stream (f32)
  __shared__ _Float16 hbuf[SEQ][DM];                 //  8 KB, LN output (WMMA A source)
  __shared__ __align__(16) unsigned char scratch[38912];
  _Float16* qkv = (_Float16*)scratch;                // [32][384] f16  (24 KB)
  float*    sb  = (float*)(scratch + 24576);         // [32][32] scores f32 (4 KB)
  _Float16* pb  = (_Float16*)(scratch + 28672);      // [32][32] probs f16 (2 KB)
  _Float16* ab  = (_Float16*)(scratch + 30720);      // [32][128] attn f16 (8 KB)
  _Float16* fb  = (_Float16*)scratch;                // [32][512] FFN mid f16 (32 KB, reuses)

  const int t = threadIdx.x, lane = t & 31, w = t >> 5;
  const int node = blockIdx.x;
  const size_t base = (size_t)node * (SEQ * DM);

  // ---- TDM async tile load: x[node] (f32, 32 rows x 128) -> LDS xbuf ----
  if (w == 0) {
    tdm_load_tile((unsigned)(size_t)(void*)&xbuf[0][0], x + base,
                  /*4B*/2u, DM, SEQ, KN * SEQ);
  }
  // warm L2 with the packed f16 weight block (425984 B) while TDM runs
  {
    const char* wb = (const char*)Wqkv;
    __builtin_prefetch(wb + t * 1664, 0, 2);
    __builtin_prefetch(wb + t * 1664 + 832, 0, 2);
  }
  if (w == 0) __builtin_amdgcn_s_wait_tensorcnt(0);
  __syncthreads();

  ln_rows(xbuf, hbuf, ln1_g, ln1_b, t);
  __syncthreads();

  // ---- QKV: [32,128] @ [128,384] ; 2x24 = 48 tiles over 8 waves ----
  for (int tile = w; tile < 48; tile += 8) {
    int m0 = (tile & 1) * 16, n0 = (tile >> 1) * 16;
    v8f c = {};
#pragma unroll
    for (int k0 = 0; k0 < DM; k0 += 32) {
      v16h a = ldA(&hbuf[0][0], DM, m0, k0, lane);
      v16h b = ldB_nk(Wqkv, DM, n0, k0, lane);
      c = WMMA_F16(a, b, c);
    }
    int n = n0 + (lane & 15), mh = (lane >> 4) * 8;
    float bias = qkv_b[n];
#pragma unroll
    for (int r = 0; r < 8; ++r) qkv[(m0 + mh + r) * 384 + n] = (_Float16)(c[r] + bias);
  }
  __syncthreads();

  // ---- scores = q k^T / sqrt(128), 4 entries per thread, 16B LDS loads ----
#pragma unroll
  for (int p = 0; p < 4; ++p) {
    int e = t + 256 * p;
    int i = e >> 5, j = e & 31;
    const v8h* qi = (const v8h*)(qkv + i * 384);
    const v8h* kj = (const v8h*)(qkv + j * 384 + DM);
    float acc = 0.f;
#pragma unroll
    for (int d = 0; d < DM / 8; ++d) {
      v8h qa = qi[d], kb = kj[d];
#pragma unroll
      for (int u = 0; u < 8; ++u) acc += (float)qa[u] * (float)kb[u];
    }
    sb[i * 32 + j] = acc * 0.08838834764831845f;
  }
  __syncthreads();

  // ---- row softmax (8 threads/row, 4 cols each) ----
  {
    int row = t >> 3, sub = t & 7;
    float mx = -1e30f;
#pragma unroll
    for (int i = 0; i < 4; ++i) mx = fmaxf(mx, sb[row * 32 + sub * 4 + i]);
    for (int m = 4; m; m >>= 1) mx = fmaxf(mx, __shfl_xor(mx, m, 8));
    float s = 0.f, ex[4];
#pragma unroll
    for (int i = 0; i < 4; ++i) { ex[i] = __expf(sb[row * 32 + sub * 4 + i] - mx); s += ex[i]; }
    for (int m = 4; m; m >>= 1) s += __shfl_xor(s, m, 8);
    float inv = 1.f / s;
#pragma unroll
    for (int i = 0; i < 4; ++i) pb[row * 32 + sub * 4 + i] = (_Float16)(ex[i] * inv);
  }
  __syncthreads();

  // ---- attn = P @ V : [32,32] @ [32,128], K=32 single step ----
  for (int tile = w; tile < 16; tile += 8) {
    int m0 = (tile & 1) * 16, n0 = (tile >> 1) * 16;
    v16h a = ldA(pb, 32, m0, 0, lane);
    v16h b = ldB_kn(qkv + 2 * DM, 384, 0, n0, lane);
    v8f c = {};
    c = WMMA_F16(a, b, c);
    int n = n0 + (lane & 15), mh = (lane >> 4) * 8;
#pragma unroll
    for (int r = 0; r < 8; ++r) ab[(m0 + mh + r) * DM + n] = (_Float16)c[r];
  }
  __syncthreads();

  // ---- out-proj + residual into xbuf ----
  for (int tile = w; tile < 16; tile += 8) {
    int m0 = (tile & 1) * 16, n0 = (tile >> 1) * 16;
    v8f c = {};
#pragma unroll
    for (int k0 = 0; k0 < DM; k0 += 32) {
      v16h a = ldA(ab, DM, m0, k0, lane);
      v16h b = ldB_nk(Wout, DM, n0, k0, lane);
      c = WMMA_F16(a, b, c);
    }
    int n = n0 + (lane & 15), mh = (lane >> 4) * 8;
    float bias = out_b[n];
#pragma unroll
    for (int r = 0; r < 8; ++r) xbuf[m0 + mh + r][n] += c[r] + bias;
  }
  __syncthreads();

  ln_rows(xbuf, hbuf, ln2_g, ln2_b, t);
  __syncthreads();

  // ---- FFN1 + exact GELU: [32,128] @ [128,512] ; 2x32 = 64 tiles ----
  for (int tile = w; tile < 64; tile += 8) {
    int m0 = (tile & 1) * 16, n0 = (tile >> 1) * 16;
    v8f c = {};
#pragma unroll
    for (int k0 = 0; k0 < DM; k0 += 32) {
      v16h a = ldA(&hbuf[0][0], DM, m0, k0, lane);
      v16h b = ldB_nk(Wff1, DM, n0, k0, lane);
      c = WMMA_F16(a, b, c);
    }
    int n = n0 + (lane & 15), mh = (lane >> 4) * 8;
    float bias = ff1_b[n];
#pragma unroll
    for (int r = 0; r < 8; ++r) {
      float v = c[r] + bias;
      v = 0.5f * v * (1.f + erff(v * 0.70710678118654752f));
      fb[(m0 + mh + r) * DFF + n] = (_Float16)v;
    }
  }
  __syncthreads();

  // ---- FFN2 + residual -> xt (f16, global) : [32,512] @ [512,128] ----
  for (int tile = w; tile < 16; tile += 8) {
    int m0 = (tile & 1) * 16, n0 = (tile >> 1) * 16;
    v8f c = {};
#pragma unroll
    for (int k0 = 0; k0 < DFF; k0 += 32) {
      v16h a = ldA(fb, DFF, m0, k0, lane);
      v16h b = ldB_nk(Wff2, DFF, n0, k0, lane);
      c = WMMA_F16(a, b, c);
    }
    int n = n0 + (lane & 15), mh = (lane >> 4) * 8;
    float bias = ff2_b[n];
#pragma unroll
    for (int r = 0; r < 8; ++r) {
      int m = m0 + mh + r;
      xt[base + m * DM + n] = (_Float16)(xbuf[m][n] + c[r] + bias);
    }
  }
}

// ============ Kernel 3: GAT linear transform + attention logit dots ============
__launch_bounds__(256)
__global__ void k_gat_h(const _Float16* __restrict__ xt, const _Float16* __restrict__ Wgat,
                        const float* __restrict__ att_src, const float* __restrict__ att_dst,
                        _Float16* __restrict__ hg, float* __restrict__ a_s, float* __restrict__ a_d) {
  __shared__ _Float16 xin[SEQ][DM];
  __shared__ float    hout[SEQ][DM];
  const int t = threadIdx.x, lane = t & 31, w = t >> 5;
  const int node = blockIdx.x;
  const size_t base = (size_t)node * (SEQ * DM);

  // ---- TDM async tile load: xt[node] (f16, 32 rows x 128) -> LDS xin ----
  if (w == 0) {
    tdm_load_tile((unsigned)(size_t)(void*)&xin[0][0], xt + base,
                  /*2B*/1u, DM, SEQ, KN * SEQ);
    __builtin_amdgcn_s_wait_tensorcnt(0);
  }
  __syncthreads();

  for (int tile = w; tile < 16; tile += 8) {
    int m0 = (tile & 1) * 16, n0 = (tile >> 1) * 16;
    v8f c = {};
#pragma unroll
    for (int k0 = 0; k0 < DM; k0 += 32) {
      v16h a = ldA(&xin[0][0], DM, m0, k0, lane);
      v16h b = ldB_nk(Wgat, DM, n0, k0, lane);
      c = WMMA_F16(a, b, c);
    }
    int n = n0 + (lane & 15), mh = (lane >> 4) * 8;
#pragma unroll
    for (int r = 0; r < 8; ++r) {
      int m = m0 + mh + r;
      hout[m][n] = c[r];
      hg[base + m * DM + n] = (_Float16)c[r];
    }
  }
  __syncthreads();

  {
    int row = t >> 3, sub = t & 7;
    float ss = 0.f, sd = 0.f;
#pragma unroll
    for (int i = 0; i < 16; ++i) {
      int c = sub * 16 + i;
      float h = hout[row][c];
      ss += h * att_src[c];
      sd += h * att_dst[c];
    }
    for (int m = 4; m; m >>= 1) { ss += __shfl_xor(ss, m, 8); sd += __shfl_xor(sd, m, 8); }
    if (sub == 0) { a_s[node * SEQ + row] = ss; a_d[node * SEQ + row] = sd; }
  }
}

// ============ Kernel 4: edge logits (leaky relu) + segment max via ordered-uint atomicMax ====
__global__ void k_edge_logit(const long long* __restrict__ ei,
                             const float* __restrict__ a_s, const float* __restrict__ a_d,
                             float* __restrict__ ev, unsigned* __restrict__ mkey) {
  int idx = blockIdx.x * 256 + threadIdx.x;
  if (idx >= NET * SEQ) return;
  int e = idx >> 5, l = idx & 31;
  int s, d;
  if (e < NE) { s = (int)ei[e]; d = (int)ei[NE + e]; } else { s = d = e - NE; }
  float v = a_s[s * SEQ + l] + a_d[d * SEQ + l];
  v = (v > 0.f) ? v : 0.2f * v;
  ev[idx] = v;
  atomicMax(&mkey[d * SEQ + l], fenc(v));
}

// ============ Kernel 5: exp(e - max) + segment sum ============
__global__ void k_edge_expsum(const long long* __restrict__ ei,
                              float* __restrict__ ev, const unsigned* __restrict__ mkey,
                              float* __restrict__ denom) {
  int idx = blockIdx.x * 256 + threadIdx.x;
  if (idx >= NET * SEQ) return;
  int e = idx >> 5, l = idx & 31;
  int d;
  if (e < NE) { d = (int)ei[NE + e]; } else { d = e - NE; }
  float m = fdec(mkey[d * SEQ + l]);
  float x = __expf(ev[idx] - m);
  ev[idx] = x;
  atomicAdd(&denom[d * SEQ + l], x);
}

// ============ Kernel 6: out[dst] += alpha * h[src], one wave per (edge, l) row ============
__launch_bounds__(256)
__global__ void k_edge_agg(const long long* __restrict__ ei,
                           const float* __restrict__ ev, const float* __restrict__ denom,
                           const _Float16* __restrict__ hg, float* __restrict__ out) {
  int pair = blockIdx.x * 8 + (threadIdx.x >> 5);
  if (pair >= NET * SEQ) return;
  int e = pair >> 5, l = pair & 31;
  int s, d;
  if (e < NE) { s = (int)ei[e]; d = (int)ei[NE + e]; } else { s = d = e - NE; }
  float alpha = ev[pair] / (denom[d * SEQ + l] + 1e-16f);
  int lane = threadIdx.x & 31;
  v4h hv = *(const v4h*)(hg + ((size_t)s * SEQ + l) * DM + lane * 4);
  float* op = out + ((size_t)d * SEQ + l) * DM + lane * 4;
#pragma unroll
  for (int i = 0; i < 4; ++i) atomicAdd(op + i, (float)hv[i] * alpha);
}

// ============ host launcher ============
extern "C" void kernel_launch(void* const* d_in, const int* in_sizes, int n_in,
                              void* d_out, int out_size, void* d_ws, size_t ws_size,
                              hipStream_t stream) {
  (void)in_sizes; (void)n_in; (void)out_size; (void)ws_size;

  const float*     x       = (const float*)d_in[0];
  const long long* ei      = (const long long*)d_in[1];
  const float*     ln1_g   = (const float*)d_in[2];
  const float*     ln1_b   = (const float*)d_in[3];
  const float*     qkv_w   = (const float*)d_in[4];
  const float*     qkv_b   = (const float*)d_in[5];
  const float*     out_w   = (const float*)d_in[6];
  const float*     out_b   = (const float*)d_in[7];
  const float*     ln2_g   = (const float*)d_in[8];
  const float*     ln2_b   = (const float*)d_in[9];
  const float*     ff1_w   = (const float*)d_in[10];
  const float*     ff1_b   = (const float*)d_in[11];
  const float*     ff2_w   = (const float*)d_in[12];
  const float*     ff2_b   = (const float*)d_in[13];
  const float*     gat_w   = (const float*)d_in[14];
  const float*     att_src = (const float*)d_in[15];
  const float*     att_dst = (const float*)d_in[16];
  const float*     gat_b   = (const float*)d_in[17];
  float* out = (float*)d_out;

  // workspace carve-up
  _Float16* xt   = (_Float16*)d_ws;                        // [KN*SEQ*DM] f16
  _Float16* hg   = xt + (size_t)KN * SEQ * DM;             // [KN*SEQ*DM] f16
  float*    a_s  = (float*)(hg + (size_t)KN * SEQ * DM);   // [KN*SEQ]
  float*    a_d  = a_s + KN * SEQ;                         // [KN*SEQ]
  unsigned* mkey = (unsigned*)(a_d + KN * SEQ);            // [KN*SEQ]
  float*    den  = (float*)(mkey + KN * SEQ);              // [KN*SEQ]
  float*    ev   = den + KN * SEQ;                         // [NET*SEQ]
  _Float16* w16  = (_Float16*)(ev + (size_t)NET * SEQ);    // 212992 halves
  const _Float16* Wqkv = w16;
  const _Float16* Wout = w16 + 49152;
  const _Float16* Wff1 = w16 + 65536;
  const _Float16* Wff2 = w16 + 131072;
  const _Float16* Wgat = w16 + 196608;

  k_prep<<<(212992 + 255) / 256, 256, 0, stream>>>(qkv_w, out_w, ff1_w, ff2_w, gat_w, w16);
  k_init<<<(KN * SEQ * DM + 255) / 256, 256, 0, stream>>>(out, gat_b, mkey, den);
  k_temporal<<<KN, 256, 0, stream>>>(x, ln1_g, ln1_b, qkv_b, out_b, ln2_g, ln2_b,
                                     ff1_b, ff2_b, Wqkv, Wout, Wff1, Wff2, xt);
  k_gat_h<<<KN, 256, 0, stream>>>(xt, Wgat, att_src, att_dst, hg, a_s, a_d);

  const int nel = NET * SEQ;
  k_edge_logit<<<(nel + 255) / 256, 256, 0, stream>>>(ei, a_s, a_d, ev, mkey);
  k_edge_expsum<<<(nel + 255) / 256, 256, 0, stream>>>(ei, ev, mkey, den);
  k_edge_agg<<<(nel + 7) / 8, 256, 0, stream>>>(ei, ev, den, hg, out);
}
